// DeformBlock_64218351010352
// MI455X (gfx1250) — compile-verified, matrix operand order checked by reading, not run on hardware
//
#include <hip/hip_runtime.h>
#include <math.h>

// ---------------------------------------------------------------------------
// Conditional Neural-ODE point deformation, persistent per-tile RK4 on WMMA.
// 4 waves / 64 points per block. Activations (64x512 bf16) ping-pong between
// two LDS buffers; weight B-panels (16x512 bf16 = 16KB) are staged into a
// single shared LDS buffer (4x cut in L2 weight traffic), with the next panel
// prefetched into registers so the global latency hides under the WMMA chain.
// Block LDS = 147KB < 160KB so two blocks can co-reside per WGP (2 waves/SIMD).
// GEMMs: v_wmma_f32_16x16x32_bf16, f32 accumulate, two parallel K-chains.
// ---------------------------------------------------------------------------

typedef __attribute__((ext_vector_type(16))) __bf16 v16bf;
typedef __attribute__((ext_vector_type(8)))  __bf16 v8bf;
typedef __attribute__((ext_vector_type(8)))  float  v8f;

#define AS1 __attribute__((address_space(1)))
typedef const __bf16 AS1* gbf;     // global bf16
typedef const float  AS1* gfl;     // global f32
typedef const v8bf   AS1* gv8;     // global 16B bf16 vector

// generic -> global addrspace cast (bit-identical on amdgcn)
__device__ __forceinline__ gbf g_bf(const void* p) { return (gbf)(size_t)p; }
__device__ __forceinline__ gfl g_fl(const void* p) { return (gfl)(size_t)p; }

#define HDIM   512
#define NPB    64            // points (rows) per block = 4 waves x 16-row strip
#define NTHR   128           // 4 waves
#define DT     0.05f         // TIME / N_STEPS = 0.2 / 4
#define PANEL  (16 * HDIM)   // bf16 elements in one 16-col weight panel

// ---- workspace layout (bytes) ----
#define OFF_W2_0 0u
#define OFF_W3_0 524288u
#define OFF_W2_1 1048576u
#define OFF_W3_1 1572864u
#define OFF_W1_0 2097152u
#define OFF_W1_1 2101248u
#define OFF_W4_0 2105344u
#define OFF_W4_1 2109440u
#define OFF_SF   2113536u    // float sf[2][4][512]

// hA + hB (2 x 64KB) + single B panel (16KB) + RK4 state (3KB) = 147KB
#define LDS_BYTES (2 * NPB * HDIM * 2 + PANEL * 2 + 4 * NPB * 3 * 4)

__device__ __forceinline__ v16bf mk16(v8bf lo, v8bf hi) {
  v16bf r;
#pragma unroll
  for (int i = 0; i < 8; ++i) { r[i] = lo[i]; r[i + 8] = hi[i]; }
  return r;
}

// Block-cooperative: out[m,n] = relu(sum_k hSrc[m,k]*W[n,k] + bias[n]) + hSrc[m,n]
// hSrc/hDst: LDS 64x512 bf16 row-major (wave w owns rows 16w..16w+15).
// W: global bf16 [512 out][512 in]. Bbuf: LDS 16x512 shared weight panel.
__device__ __forceinline__ void block_gemm_res(
    const __bf16* hSrc, __bf16* hDst, gbf W, gfl bias,
    __bf16* Bbuf, int tid) {
  const int lane  = tid & 31;
  const int wave  = tid >> 5;
  const int mr    = lane & 15;               // A row / B col within 16-tile
  const int koff  = (lane & 16) ? 8 : 0;     // K sub-offset per lane half
  const int mbase = (lane & 16) ? 8 : 0;     // C/D row base per lane half
  const __bf16* hS = hSrc + wave * 16 * HDIM;
  __bf16*       hD = hDst + wave * 16 * HDIM;

  // Hoist this wave's entire 16x512 A panel (bf16) into 128 VGPRs (ds_load).
  v16bf afrag[16];
#pragma unroll
  for (int kc = 0; kc < 16; ++kc) {
    v8bf lo = *(const v8bf*)(hS + mr * HDIM + kc * 32 + koff);
    v8bf hi = *(const v8bf*)(hS + mr * HDIM + kc * 32 + 16 + koff);
    afrag[kc] = mk16(lo, hi);
  }

  // Stage panel 0 cooperatively: 16KB / 128 threads = 128B (8 x b128) each.
  {
    gbf src = W + tid * 64;
    __bf16* dst = Bbuf + tid * 64;
#pragma unroll
    for (int i = 0; i < 8; ++i) ((v8bf*)dst)[i] = ((gv8)src)[i];
  }
  __syncthreads();

#pragma unroll 1
  for (int nt = 0; nt < 32; ++nt) {          // 32 column tiles of 16
    const int n0 = nt * 16;

    // Prefetch next panel into registers (latency hides under the WMMAs).
    v8bf st[8];
    if (nt < 31) {
      gbf nsrc = W + (nt + 1) * PANEL + tid * 64;
#pragma unroll
      for (int i = 0; i < 8; ++i) st[i] = ((gv8)nsrc)[i];
    }

    const __bf16* wp = Bbuf + mr * HDIM + koff;  // panel row n, K-contiguous
    v8f acc0 = {0.f, 0.f, 0.f, 0.f, 0.f, 0.f, 0.f, 0.f};
    v8f acc1 = {0.f, 0.f, 0.f, 0.f, 0.f, 0.f, 0.f, 0.f};
#pragma unroll
    for (int kp = 0; kp < 8; ++kp) {         // K = 512, two parallel chains
      const int kcA = 2 * kp, kcB = 2 * kp + 1;
      v8bf loA = *(const v8bf*)(wp + kcA * 32);
      v8bf hiA = *(const v8bf*)(wp + kcA * 32 + 16);
      v8bf loB = *(const v8bf*)(wp + kcB * 32);
      v8bf hiB = *(const v8bf*)(wp + kcB * 32 + 16);
      acc0 = __builtin_amdgcn_wmma_f32_16x16x32_bf16(
          false, afrag[kcA], false, mk16(loA, hiA), (short)0, acc0, false, false);
      acc1 = __builtin_amdgcn_wmma_f32_16x16x32_bf16(
          false, afrag[kcB], false, mk16(loB, hiB), (short)0, acc1, false, false);
    }
    const float bn = bias[n0 + mr];
#pragma unroll
    for (int j = 0; j < 8; ++j) {            // epilogue: bias+relu+residual
      const int m = mbase + j;
      float v = acc0[j] + acc1[j] + bn;
      v = v > 0.f ? v : 0.f;
      v += (float)hS[m * HDIM + n0 + mr];
      hD[m * HDIM + n0 + mr] = (__bf16)v;
    }

    __syncthreads();                         // all waves done reading Bbuf
    if (nt < 31) {                           // commit prefetched panel
      __bf16* nd = Bbuf + tid * 64;
#pragma unroll
      for (int i = 0; i < 8; ++i) ((v8bf*)nd)[i] = st[i];
      __syncthreads();                       // panel visible to all waves
    }
  }
}

__global__ __launch_bounds__(NTHR) void node_rk4_kernel(
    const float* __restrict__ x, float* __restrict__ out,
    const unsigned short* W1_0s, const unsigned short* W2_0s,
    const unsigned short* W3_0s, const unsigned short* W4_0s,
    const unsigned short* W1_1s, const unsigned short* W2_1s,
    const unsigned short* W3_1s, const unsigned short* W4_1s,
    const float* b1_0, const float* b2_0, const float* b3_0, const float* b4_0,
    const float* b1_1, const float* b2_1, const float* b3_1, const float* b4_1,
    const float* __restrict__ sf_all) {
  extern __shared__ char smem[];
  __bf16* hA    = (__bf16*)smem;                 // 64x512 bf16 = 64KB
  __bf16* hB    = hA + NPB * HDIM;               // 64KB
  __bf16* Bbuf  = hB + NPB * HDIM;               // 16KB weight panel
  float*  p_cur = (float*)(Bbuf + PANEL);        // 64x3 each
  float*  p_evl = p_cur + NPB * 3;
  float*  k_sum = p_evl + NPB * 3;
  float*  k_cur = k_sum + NPB * 3;

  const int tid   = threadIdx.x;
  const int row0  = blockIdx.x * NPB;
  const int batch = row0 >> 12;                  // 4096 points per batch

  for (int i = tid; i < NPB * 3; i += NTHR) {
    float v = x[row0 * 3 + i];
    p_cur[i] = v; p_evl[i] = v;
  }

#pragma unroll 1
  for (int f = 0; f < 2; ++f) {                  // two sequential NODE blocks
    gbf W1 = g_bf(f ? (const void*)W1_1s : (const void*)W1_0s);
    gbf W2 = g_bf(f ? (const void*)W2_1s : (const void*)W2_0s);
    gbf W3 = g_bf(f ? (const void*)W3_1s : (const void*)W3_0s);
    gbf W4 = g_bf(f ? (const void*)W4_1s : (const void*)W4_0s);
    gfl b1 = g_fl(f ? b1_1 : b1_0);
    gfl b2 = g_fl(f ? b2_1 : b2_0);
    gfl b3 = g_fl(f ? b3_1 : b3_0);
    gfl b4 = g_fl(f ? b4_1 : b4_0);
    gfl sf = g_fl(sf_all) + (f * 4 + batch) * HDIM;

#pragma unroll 1
    for (int step = 0; step < 4; ++step) {       // RK4 steps
      for (int i = tid; i < NPB * 3; i += NTHR) k_sum[i] = 0.f;
#pragma unroll 1
      for (int e = 0; e < 4; ++e) {              // k1..k4 evals
        __syncthreads();
        // ---- layer1: pf = relu(p*W1^T+b1), gated by sf -> hA (bf16) ----
        for (int idx = tid; idx < NPB * HDIM; idx += NTHR) {
          int m = idx >> 9, n = idx & (HDIM - 1);
          float p0 = p_evl[m * 3 + 0], p1 = p_evl[m * 3 + 1],
                p2 = p_evl[m * 3 + 2];
          float v = fmaf(p0, (float)W1[n * 3 + 0],
                    fmaf(p1, (float)W1[n * 3 + 1],
                    fmaf(p2, (float)W1[n * 3 + 2], b1[n])));
          v = v > 0.f ? v : 0.f;
          v *= sf[n];
          hA[idx] = (__bf16)v;
        }
        __syncthreads();
        block_gemm_res(hA, hB, W2, b2, Bbuf, tid); // residual stage 1
        __syncthreads();
        block_gemm_res(hB, hA, W3, b3, Bbuf, tid); // residual stage 2
        __syncthreads();
        // ---- layer4: k = tanh(h*W4^T + b4), 192 dots of length 512 ----
        for (int t = tid; t < NPB * 3; t += NTHR) {
          int m = t / 3, c = t - m * 3;
          const __bf16* hr = hA + m * HDIM;
          gbf wr = W4 + c * HDIM;
          float s = 0.f;
#pragma unroll 2
          for (int k = 0; k < HDIM; k += 8) {
            v8bf hv = *(const v8bf*)(hr + k);
            v8bf wv = *(gv8)(wr + k);
#pragma unroll
            for (int j = 0; j < 8; ++j) s = fmaf((float)hv[j], (float)wv[j], s);
          }
          k_cur[t] = tanhf(s + b4[c]);
        }
        __syncthreads();
        const float wgt = (e == 1 || e == 2) ? 2.f : 1.f;
        const float ac  = (e == 2) ? DT : 0.5f * DT;
        for (int i = tid; i < NPB * 3; i += NTHR) {
          float k = k_cur[i];
          k_sum[i] += wgt * k;
          if (e < 3) p_evl[i] = p_cur[i] + ac * k;
        }
      }
      __syncthreads();
      for (int i = tid; i < NPB * 3; i += NTHR) {
        float v = p_cur[i] + (DT / 6.f) * k_sum[i];
        p_cur[i] = v; p_evl[i] = v;
      }
    }
  }
  __syncthreads();
  for (int i = tid; i < NPB * 3; i += NTHR) out[row0 * 3 + i] = p_cur[i];
}

// ---------------------------------------------------------------------------
// Prologue: f32 -> bf16 weight conversion; sf = tanh(code @ cond_w^T + cond_b)
// ---------------------------------------------------------------------------
__global__ void cvt_bf16_kernel(const float* __restrict__ src,
                                unsigned short* __restrict__ dst, int n) {
  int i = blockIdx.x * blockDim.x + threadIdx.x;
  if (i < n) ((__bf16*)dst)[i] = (__bf16)src[i];
}

__global__ void sf_kernel(const float* __restrict__ code,
                          const float* cw0, const float* cb0,
                          const float* cw1, const float* cb1,
                          float* __restrict__ sf) {
  int tid = blockIdx.x * blockDim.x + threadIdx.x;   // 2 funcs * 4 batch * 512
  int f = tid >> 11, r = tid & 2047, b = r >> 9, n = r & 511;
  gfl cw = g_fl(f ? cw1 : cw0);
  gfl cb = g_fl(f ? cb1 : cb0);
  gfl c = g_fl(code) + b * HDIM;
  gfl w = cw + n * HDIM;
  float s = cb[n];
  for (int k = 0; k < HDIM; k += 4)
    s = fmaf(c[k], w[k],
        fmaf(c[k + 1], w[k + 1],
        fmaf(c[k + 2], w[k + 2],
        fmaf(c[k + 3], w[k + 3], s))));
  sf[tid] = tanhf(s);
}

extern "C" void kernel_launch(void* const* d_in, const int* in_sizes, int n_in,
                              void* d_out, int out_size, void* d_ws,
                              size_t ws_size, hipStream_t stream) {
  const float* code    = (const float*)d_in[0];
  const float* x       = (const float*)d_in[1];
  const float* f1_l1_w = (const float*)d_in[2];
  const float* f1_l1_b = (const float*)d_in[3];
  const float* f1_l2_w = (const float*)d_in[4];
  const float* f1_l2_b = (const float*)d_in[5];
  const float* f1_l3_w = (const float*)d_in[6];
  const float* f1_l3_b = (const float*)d_in[7];
  const float* f1_l4_w = (const float*)d_in[8];
  const float* f1_l4_b = (const float*)d_in[9];
  const float* f1_cw   = (const float*)d_in[10];
  const float* f1_cb   = (const float*)d_in[11];
  const float* f2_l1_w = (const float*)d_in[12];
  const float* f2_l1_b = (const float*)d_in[13];
  const float* f2_l2_w = (const float*)d_in[14];
  const float* f2_l2_b = (const float*)d_in[15];
  const float* f2_l3_w = (const float*)d_in[16];
  const float* f2_l3_b = (const float*)d_in[17];
  const float* f2_l4_w = (const float*)d_in[18];
  const float* f2_l4_b = (const float*)d_in[19];
  const float* f2_cw   = (const float*)d_in[20];
  const float* f2_cb   = (const float*)d_in[21];

  char* ws = (char*)d_ws;
  unsigned short* W2_0 = (unsigned short*)(ws + OFF_W2_0);
  unsigned short* W3_0 = (unsigned short*)(ws + OFF_W3_0);
  unsigned short* W2_1 = (unsigned short*)(ws + OFF_W2_1);
  unsigned short* W3_1 = (unsigned short*)(ws + OFF_W3_1);
  unsigned short* W1_0 = (unsigned short*)(ws + OFF_W1_0);
  unsigned short* W1_1 = (unsigned short*)(ws + OFF_W1_1);
  unsigned short* W4_0 = (unsigned short*)(ws + OFF_W4_0);
  unsigned short* W4_1 = (unsigned short*)(ws + OFF_W4_1);
  float* sf = (float*)(ws + OFF_SF);

  const int nHH = HDIM * HDIM;   // 262144
  const int nSm = 3 * HDIM;      // 1536
  cvt_bf16_kernel<<<(nHH + 255) / 256, 256, 0, stream>>>(f1_l2_w, W2_0, nHH);
  cvt_bf16_kernel<<<(nHH + 255) / 256, 256, 0, stream>>>(f1_l3_w, W3_0, nHH);
  cvt_bf16_kernel<<<(nHH + 255) / 256, 256, 0, stream>>>(f2_l2_w, W2_1, nHH);
  cvt_bf16_kernel<<<(nHH + 255) / 256, 256, 0, stream>>>(f2_l3_w, W3_1, nHH);
  cvt_bf16_kernel<<<(nSm + 255) / 256, 256, 0, stream>>>(f1_l1_w, W1_0, nSm);
  cvt_bf16_kernel<<<(nSm + 255) / 256, 256, 0, stream>>>(f2_l1_w, W1_1, nSm);
  cvt_bf16_kernel<<<(nSm + 255) / 256, 256, 0, stream>>>(f1_l4_w, W4_0, nSm);
  cvt_bf16_kernel<<<(nSm + 255) / 256, 256, 0, stream>>>(f2_l4_w, W4_1, nSm);
  sf_kernel<<<16, 256, 0, stream>>>(code, f1_cw, f1_cb, f2_cw, f2_cb, sf);

  // Allow >64KB dynamic LDS (gfx1250: up to 320KB per workgroup).
  (void)hipFuncSetAttribute((const void*)node_rk4_kernel,
                            hipFuncAttributeMaxDynamicSharedMemorySize,
                            (int)LDS_BYTES);

  const int nRows = 4 * 4096;
  node_rk4_kernel<<<nRows / NPB, NTHR, LDS_BYTES, stream>>>(
      x, (float*)d_out,
      W1_0, W2_0, W3_0, W4_0, W1_1, W2_1, W3_1, W4_1,
      f1_l1_b, f1_l2_b, f1_l3_b, f1_l4_b,
      f2_l1_b, f2_l2_b, f2_l3_b, f2_l4_b,
      sf);
}
